// WindowAttention3D_65206193488028
// MI455X (gfx1250) — matrix operand order repaired and checked by reading, not compiled, not run on hardware
//
#include <hip/hip_runtime.h>
#include <hip/hip_bf16.h>

typedef _Float16 half_t;
typedef __attribute__((ext_vector_type(16))) _Float16 v16h;
typedef __attribute__((ext_vector_type(8)))  _Float16 h8;   // 16-byte copy unit
typedef __attribute__((ext_vector_type(4)))  _Float16 h4;   // 8-byte packed store
typedef __attribute__((ext_vector_type(8)))  float    v8f;

#define DIM     512
#define NHEADS  16
#define HD      32
#define NTOK    144
#define NWIN    96
#define BWIN    768
#define NPAD    160
#define MTOT    (BWIN * NTOK)   /* 110592 tokens */
#define LDA     72              /* LDS leading dim (halves): 144B, conflict-free */

// ---------------------------------------------------------------------------
// WMMA helpers (CDNA5: D = A(16x32 f16) * B(32x16 f16) + C(16x16 f32))
// ---------------------------------------------------------------------------
__device__ __forceinline__ v8f wmma_f16f32(v16h a, v16h b, v8f c) {
  return __builtin_amdgcn_wmma_f32_16x16x32_f16(
      /*neg_a=*/false, a, /*neg_b=*/false, b,
      /*c_mod=*/(short)0, c, /*reuse_a=*/false, /*reuse_b=*/false);
}

// A-matrix 16x32 f16 fragment (ISA 7.12.2): lane L holds row M=L&15;
// lanes 0-15 hold K = {0..7, 16..23}, lanes 16-31 hold K = {8..15, 24..31}.
__device__ __forceinline__ v16h load_frag_a(const half_t* src, int ld) {
  const int lane = threadIdx.x & 31;
  const half_t* row = src + (lane & 15) * ld;
  const int hs = (lane >> 4) * 8;
  v16h a;
#pragma unroll
  for (int v = 0; v < 8; ++v) {
    const int kb = (v < 4) ? (hs + v * 2) : (16 + hs + (v - 4) * 2);
    a[2 * v]     = row[kb];
    a[2 * v + 1] = row[kb + 1];
  }
  return a;
}

// B-matrix 32x16 f16 fragment: lane L holds col N=L&15; lanes 0-15 hold
// K = {0..7, 16..23}, lanes 16-31 hold K = {8..15, 24..31}. Element (k,n) from
// src[n*ldn + k*ldk].
__device__ __forceinline__ v16h load_frag_b(const half_t* src, int ldk, int ldn) {
  const int lane = threadIdx.x & 31;
  const half_t* col = src + (lane & 15) * ldn;
  const int hs = (lane >> 4) * 8;
  v16h b;
#pragma unroll
  for (int v = 0; v < 8; ++v) {
#pragma unroll
    for (int p = 0; p < 2; ++p) {
      b[2 * v + p] = col[(((v < 4) ? (hs + v * 2) : (16 + hs + (v - 4) * 2)) + p) * ldk];
    }
  }
  return b;
}

// ---------------------------------------------------------------------------
// Kernel 1: densify relative-position bias -> [H, 144, 144] f32 (L2-resident)
// ---------------------------------------------------------------------------
__global__ __launch_bounds__(256) void bias_gather(
    const float* __restrict__ table, const int* __restrict__ rel_index,
    float* __restrict__ bias_full) {
  const int idx = blockIdx.x * 256 + threadIdx.x;
  const int total = NHEADS * NTOK * NTOK;
  if (idx >= total) return;
  const int h  = idx / (NTOK * NTOK);
  const int nm = idx - h * (NTOK * NTOK);
  bias_full[idx] = table[rel_index[nm] * NHEADS + h];
}

// ---------------------------------------------------------------------------
// Kernel 2: QKV GEMM  qkv = x @ qkv_w^T + qkv_b  (f32 in, f16 out)
// Tile 128(M) x 64(N), BLOCK_K = 64; 256 thr = 8 waves (4x2), 8 WMMA / iter.
// Staging: float4 global loads -> packed 4xf16 (b64) LDS stores.
// q-columns are pre-scaled by hd^-0.5 in the epilogue (folds softmax scale).
// Output reordered to [b][3][head][144][32] f16.
// ---------------------------------------------------------------------------
__global__ __launch_bounds__(256) void qkv_gemm(
    const float* __restrict__ x, const float* __restrict__ w,
    const float* __restrict__ bias, half_t* __restrict__ out) {
  __shared__ half_t As[128][LDA];
  __shared__ half_t Bs[64][LDA];

  const int m_base = blockIdx.x * 128;
  const int n_base = blockIdx.y * 64;
  const int tid  = threadIdx.x;
  const int lane = tid & 31;
  const int wave = tid >> 5;
  const int wm = wave & 3;   // 4 waves over M
  const int wn = wave >> 2;  // 2 waves over N

  v8f acc[2][2];
#pragma unroll
  for (int i = 0; i < 2; ++i)
#pragma unroll
    for (int j = 0; j < 2; ++j)
      acc[i][j] = (v8f){0.f, 0.f, 0.f, 0.f, 0.f, 0.f, 0.f, 0.f};

  for (int kk = 0; kk < DIM; kk += 64) {
    if (kk + 64 < DIM) {  // warm next tiles toward L2
      __builtin_prefetch(&x[(size_t)(m_base + (tid >> 1)) * DIM + kk + 64], 0, 1);
      __builtin_prefetch(&w[(size_t)(n_base + (tid & 63)) * DIM + kk + 64], 0, 1);
    }
    // A: 128x64 f32 -> f16 : 2048 float4, 8 per thread
#pragma unroll
    for (int q = 0; q < 8; ++q) {
      const int linear = q * 256 + tid;
      const int r  = linear >> 4;   // 16 float4 per row
      const int c4 = (linear & 15) * 4;
      const float4 f =
          *(const float4*)&x[(size_t)(m_base + r) * DIM + kk + c4];
      const h4 hv = {(half_t)f.x, (half_t)f.y, (half_t)f.z, (half_t)f.w};
      *(h4*)&As[r][c4] = hv;
    }
    // B: 64x64 f32 -> f16 : 1024 float4, 4 per thread
#pragma unroll
    for (int q = 0; q < 4; ++q) {
      const int linear = q * 256 + tid;
      const int r  = linear >> 4;
      const int c4 = (linear & 15) * 4;
      const float4 f =
          *(const float4*)&w[(size_t)(n_base + r) * DIM + kk + c4];
      const h4 hv = {(half_t)f.x, (half_t)f.y, (half_t)f.z, (half_t)f.w};
      *(h4*)&Bs[r][c4] = hv;
    }
    __syncthreads();

#pragma unroll
    for (int ks = 0; ks < 64; ks += 32) {
      v16h af[2], bf[2];
      af[0] = load_frag_a(&As[wm * 32][ks], LDA);
      af[1] = load_frag_a(&As[wm * 32 + 16][ks], LDA);
      bf[0] = load_frag_b(&Bs[wn * 32][ks], /*ldk=*/1, /*ldn=*/LDA);
      bf[1] = load_frag_b(&Bs[wn * 32 + 16][ks], 1, LDA);
#pragma unroll
      for (int i = 0; i < 2; ++i)
#pragma unroll
        for (int j = 0; j < 2; ++j)
          acc[i][j] = wmma_f16f32(af[i], bf[j], acc[i][j]);
    }
    __syncthreads();
  }

  // Epilogue: +bias, fold attention scale into q, scatter to [b][3][h][n][d].
  const int coln = lane & 15;
  const int rowh = (lane >> 4) * 8;
#pragma unroll
  for (int i = 0; i < 2; ++i)
#pragma unroll
    for (int j = 0; j < 2; ++j) {
      const int jc = n_base + wn * 32 + j * 16 + coln;
      const int which = jc >> 9;          // 0:q 1:k 2:v
      const int rem   = jc & 511;
      const int hh = rem >> 5, dd = rem & 31;
      const float sc = (which == 0) ? 0.17677669529663687f : 1.0f;
      const float bj = bias[jc];
#pragma unroll
      for (int r = 0; r < 8; ++r) {
        const int m = m_base + wm * 32 + i * 16 + r + rowh;
        const int bw = m / NTOK;
        const int n  = m - bw * NTOK;
        const size_t dst =
            ((((size_t)bw * 3 + which) * NHEADS + hh) * NTOK + n) * HD + dd;
        out[dst] = (half_t)((acc[i][j][r] + bj) * sc);
      }
    }
}

// ---------------------------------------------------------------------------
// Kernel 3: attention, one (b, head) per workgroup. 288 thr = 9 waves,
// wave w owns rows [16w, 16w+16). S = (q*scale) K^T (9 WMMA) -> +bias +mask
// -> softmax (16-lane shfl, matches C/D layout) -> O = P V (10 WMMA,
// K padded 144->160 with zeros). Per-wave LDS bounce for C/D->A transpose.
// ---------------------------------------------------------------------------
__global__ __launch_bounds__(288) void attn_kernel(
    const half_t* __restrict__ qkv, const float* __restrict__ bias_full,
    const float* __restrict__ mask, half_t* __restrict__ o_h) {
  __shared__ half_t Qh[NTOK][HD];      //  9 KB
  __shared__ half_t Kh[NTOK][HD];      //  9 KB
  __shared__ half_t Vh[NPAD][HD];      // 10 KB (rows 144..159 zero)
  __shared__ half_t Pslab[9][16][40];  // ~11.5 KB per-wave P bounce

  const int b = blockIdx.x >> 4;
  const int h = blockIdx.x & 15;
  const int tid  = threadIdx.x;
  const int lane = tid & 31;
  const int wave = tid / 32;
  const int n0 = wave * 16;

  // 16-byte vector staging: 576 h8 per matrix, 2 per thread. Bases are
  // multiples of 4608 halves (9216 B) -> 16B aligned.
  {
    const h8* qsrc = (const h8*)(qkv + (((size_t)b * 3 + 0) * NHEADS + h) * (NTOK * HD));
    const h8* ksrc = (const h8*)(qkv + (((size_t)b * 3 + 1) * NHEADS + h) * (NTOK * HD));
    const h8* vsrc = (const h8*)(qkv + (((size_t)b * 3 + 2) * NHEADS + h) * (NTOK * HD));
    h8* Qv = (h8*)&Qh[0][0];
    h8* Kv = (h8*)&Kh[0][0];
    h8* Vv = (h8*)&Vh[0][0];
#pragma unroll
    for (int q = 0; q < 2; ++q) {
      const int i = q * 288 + tid;
      Qv[i] = qsrc[i];
      Kv[i] = ksrc[i];
      Vv[i] = vsrc[i];
    }
    if (tid < 64) {  // zero V pad rows 144..159 (64 h8)
      const h8 z = {(half_t)0, (half_t)0, (half_t)0, (half_t)0,
                    (half_t)0, (half_t)0, (half_t)0, (half_t)0};
      Vv[576 + tid] = z;
    }
  }
  __syncthreads();

  // ---- S = (q*scale) K^T : one A frag (reused), 9 B frags ----
  const v16h aQ = load_frag_a(&Qh[n0][0], HD);
  v8f acc[9];
#pragma unroll
  for (int t = 0; t < 9; ++t) {
    const v16h bK = load_frag_b(&Kh[t * 16][0], /*ldk=*/1, /*ldn=*/HD);
    const v8f z = (v8f){0.f, 0.f, 0.f, 0.f, 0.f, 0.f, 0.f, 0.f};
    acc[t] = wmma_f16f32(aQ, bK, z);
  }

  // ---- + rel-pos bias + window mask, then row softmax ----
  const float* bh = bias_full + (size_t)h * NTOK * NTOK;
  const float* mw = mask + (size_t)(b % NWIN) * NTOK * NTOK;
  const int colg = lane & 15;
  const int rowh = (lane >> 4) * 8;

  float rinv[8];
#pragma unroll
  for (int r = 0; r < 8; ++r) {
    const int n = n0 + r + rowh;
    float mx = -1e30f;
#pragma unroll
    for (int t = 0; t < 9; ++t) {
      const int m = t * 16 + colg;
      const float s = acc[t][r] + bh[n * NTOK + m] + mw[n * NTOK + m];
      acc[t][r] = s;
      mx = fmaxf(mx, s);
    }
#pragma unroll
    for (int off = 1; off < 16; off <<= 1)
      mx = fmaxf(mx, __shfl_xor(mx, off, 32));
    float sum = 0.f;
#pragma unroll
    for (int t = 0; t < 9; ++t) {
      const float e = __expf(acc[t][r] - mx);
      acc[t][r] = e;
      sum += e;
    }
#pragma unroll
    for (int off = 1; off < 16; off <<= 1)
      sum += __shfl_xor(sum, off, 32);
    rinv[r] = 1.f / sum;
  }

  // ---- O = P V over 5 K-chunks of 32 (last half-chunk zero padded) ----
  half_t(*Pw)[40] = Pslab[wave];
  v8f oacc[2];
  oacc[0] = (v8f){0.f, 0.f, 0.f, 0.f, 0.f, 0.f, 0.f, 0.f};
  oacc[1] = oacc[0];
#pragma unroll
  for (int kc = 0; kc < 5; ++kc) {
    // C/D layout -> A layout via per-wave LDS slice (same-wave DS in-order)
#pragma unroll
    for (int ht = 0; ht < 2; ++ht) {
      const int t = 2 * kc + ht;
#pragma unroll
      for (int r = 0; r < 8; ++r) {
        const half_t v = (t < 9) ? (half_t)acc[t][r] : (half_t)0.f;
        Pw[r + rowh][ht * 16 + colg] = v;
      }
    }
    const v16h aP = load_frag_a(&Pw[0][0], 40);
#pragma unroll
    for (int j = 0; j < 2; ++j) {
      const v16h bV = load_frag_b(&Vh[kc * 32][j * 16], /*ldk=*/HD, /*ldn=*/1);
      oacc[j] = wmma_f16f32(aP, bV, oacc[j]);
    }
  }

  // ---- normalize and store as [b][n][h*32+d] f16 for the proj GEMM ----
#pragma unroll
  for (int j = 0; j < 2; ++j)
#pragma unroll
    for (int r = 0; r < 8; ++r) {
      const int n = n0 + r + rowh;
      const int d = j * 16 + colg;
      const float v = oacc[j][r] * rinv[r];
      o_h[((size_t)b * NTOK + n) * DIM + h * HD + d] = (half_t)v;
    }
}

// ---------------------------------------------------------------------------
// Kernel 4: output projection  out = o @ proj_w^T + proj_b  (f16 in, f32 out)
// Same tiling as qkv_gemm; A staged with 16B h8 copies.
// ---------------------------------------------------------------------------
__global__ __launch_bounds__(256) void proj_gemm(
    const half_t* __restrict__ a, const float* __restrict__ w,
    const float* __restrict__ bias, float* __restrict__ out) {
  __shared__ half_t As[128][LDA];
  __shared__ half_t Bs[64][LDA];

  const int m_base = blockIdx.x * 128;
  const int n_base = blockIdx.y * 64;
  const int tid  = threadIdx.x;
  const int lane = tid & 31;
  const int wave = tid >> 5;
  const int wm = wave & 3;
  const int wn = wave >> 2;

  v8f acc[2][2];
#pragma unroll
  for (int i = 0; i < 2; ++i)
#pragma unroll
    for (int j = 0; j < 2; ++j)
      acc[i][j] = (v8f){0.f, 0.f, 0.f, 0.f, 0.f, 0.f, 0.f, 0.f};

  for (int kk = 0; kk < DIM; kk += 64) {
    if (kk + 64 < DIM) {
      __builtin_prefetch(&a[(size_t)(m_base + (tid >> 1)) * DIM + kk + 64], 0, 1);
      __builtin_prefetch(&w[(size_t)(n_base + (tid & 63)) * DIM + kk + 64], 0, 1);
    }
    // A: 128x64 f16, 1024 h8, 4 per thread
#pragma unroll
    for (int q = 0; q < 4; ++q) {
      const int linear = q * 256 + tid;
      const int r  = linear >> 3;   // 8 h8 per row
      const int c8 = (linear & 7) * 8;
      *(h8*)&As[r][c8] =
          *(const h8*)&a[(size_t)(m_base + r) * DIM + kk + c8];
    }
    // B: 64x64 f32 -> f16, 4 float4 per thread
#pragma unroll
    for (int q = 0; q < 4; ++q) {
      const int linear = q * 256 + tid;
      const int r  = linear >> 4;
      const int c4 = (linear & 15) * 4;
      const float4 f =
          *(const float4*)&w[(size_t)(n_base + r) * DIM + kk + c4];
      const h4 hv = {(half_t)f.x, (half_t)f.y, (half_t)f.z, (half_t)f.w};
      *(h4*)&Bs[r][c4] = hv;
    }
    __syncthreads();

#pragma unroll
    for (int ks = 0; ks < 64; ks += 32) {
      v16h af[2], bf[2];
      af[0] = load_frag_a(&As[wm * 32][ks], LDA);
      af[1] = load_frag_a(&As[wm * 32 + 16][ks], LDA);
      bf[0] = load_frag_b(&Bs[wn * 32][ks], 1, LDA);
      bf[1] = load_frag_b(&Bs[wn * 32 + 16][ks], 1, LDA);
#pragma unroll
      for (int i = 0; i < 2; ++i)
#pragma unroll
        for (int j = 0; j < 2; ++j)
          acc[i][j] = wmma_f16f32(af[i], bf[j], acc[i][j]);
    }
    __syncthreads();
  }

  const int coln = lane & 15;
  const int rowh = (lane >> 4) * 8;
#pragma unroll
  for (int i = 0; i < 2; ++i)
#pragma unroll
    for (int j = 0; j < 2; ++j) {
      const int jc = n_base + wn * 32 + j * 16 + coln;
      const float bj = bias[jc];
#pragma unroll
      for (int r = 0; r < 8; ++r) {
        const int m = m_base + wm * 32 + i * 16 + r + rowh;
        out[(size_t)m * DIM + jc] = acc[i][j][r] + bj;
      }
    }
}

// ---------------------------------------------------------------------------
// Host launcher
// ---------------------------------------------------------------------------
extern "C" void kernel_launch(void* const* d_in, const int* in_sizes, int n_in,
                              void* d_out, int out_size, void* d_ws, size_t ws_size,
                              hipStream_t stream) {
  const float* x          = (const float*)d_in[0];
  const float* mask       = (const float*)d_in[1];
  const float* qkv_w      = (const float*)d_in[2];
  const float* qkv_b      = (const float*)d_in[3];
  const float* proj_w     = (const float*)d_in[4];
  const float* proj_b     = (const float*)d_in[5];
  const float* bias_table = (const float*)d_in[6];
  const int*   rel_index  = (const int*)d_in[7];
  float* out = (float*)d_out;

  // Workspace layout (f16 QKV, f16 attn-out, f32 dense bias)
  const size_t qkv_elems = (size_t)BWIN * 3 * NHEADS * NTOK * HD;  // 169.9M f16
  const size_t o_elems   = (size_t)MTOT * DIM;                     //  56.6M f16
  half_t* qkv_h     = (half_t*)d_ws;
  half_t* o_h       = qkv_h + qkv_elems;
  float*  bias_full = (float*)(o_h + o_elems);

  {
    const int total = NHEADS * NTOK * NTOK;
    bias_gather<<<dim3((total + 255) / 256), dim3(256), 0, stream>>>(
        bias_table, rel_index, bias_full);
  }
  qkv_gemm<<<dim3(MTOT / 128, (3 * DIM) / 64), dim3(256), 0, stream>>>(
      x, qkv_w, qkv_b, qkv_h);
  attn_kernel<<<dim3(BWIN * NHEADS), dim3(288), 0, stream>>>(
      qkv_h, bias_full, mask, o_h);
  proj_gemm<<<dim3(MTOT / 128, DIM / 64), dim3(256), 0, stream>>>(
      o_h, proj_w, proj_b, out);
}
